// Attention_87308095193213
// MI455X (gfx1250) — matrix operand order repaired and checked by reading, not compiled, not run on hardware
//
#include <hip/hip_runtime.h>

// Problem constants (match reference)
#define NB   2
#define LQ   1024
#define DMODEL 4544
#define NHEAD 71
#define DKV  64
#define NQKV (DMODEL + 2*DKV)   // 4672 = 73*64
#define MROWS (NB*LQ)           // 2048

typedef __attribute__((ext_vector_type(16))) __bf16 v16bf;
typedef __attribute__((ext_vector_type(8)))  float  v8f;
typedef __attribute__((ext_vector_type(4)))  unsigned int v4u;
typedef __attribute__((ext_vector_type(4)))  unsigned int u32x4;
typedef __attribute__((ext_vector_type(4)))  int i32x4;
typedef __attribute__((ext_vector_type(8)))  int i32x8;

union FragBF {
    v16bf bf;
    v4u u2[2];
    unsigned short s[16];
};

__device__ __forceinline__ unsigned short f2bf(float f) {
    unsigned int u = __float_as_uint(f);
    u += 0x7FFFu + ((u >> 16) & 1u);   // round-to-nearest-even
    return (unsigned short)(u >> 16);
}

// ---------------------------------------------------------------- convert ---
__global__ void cvt_f32_bf16(const float* __restrict__ src,
                             unsigned short* __restrict__ dst, size_t n) {
    size_t i = (size_t)blockIdx.x * blockDim.x + threadIdx.x;
    if (i < n) dst[i] = f2bf(src[i]);
}

// ----------------------------------------------------------- TDM support ---
#if defined(__has_builtin)
#if __has_builtin(__builtin_amdgcn_tensor_load_to_lds) && \
    __has_builtin(__builtin_amdgcn_s_wait_tensorcnt)
#define USE_TDM 1
#endif
#endif

#ifdef USE_TDM
// Issue one TDM 2D tile load: bf16 tile (tile_w x tile_h elements) from a
// row-major tensor with row stride `stride` elements into LDS at lds_off.
// D# layout per CDNA5 ISA ch.8 (group0: control/lds/global addr, group1:
// dims/strides). 2D -> groups 2/3 zero (VADDR2/3 semantics: unused dims).
__device__ __forceinline__ void tdm_load_tile_bf16(
    unsigned int lds_off, const unsigned short* g,
    unsigned int tile_w, unsigned int tile_h,
    unsigned int tensor_w, unsigned int tensor_h, unsigned int stride) {
    unsigned long long ga = (unsigned long long)g;
    u32x4 g0 = {};
    g0[0] = 1u;                                        // count=1 (valid D#)
    g0[1] = lds_off;                                   // lds_addr (bytes)
    g0[2] = (unsigned int)(ga & 0xFFFFFFFFu);          // global_addr[31:0]
    g0[3] = (unsigned int)((ga >> 32) & 0x01FFFFFFu)   // global_addr[56:32]
            | (2u << 30);                              // type=2 ("image")
    i32x8 g1 = {};
    g1[0] = (int)(1u << 16);                           // data_size=1 (2 bytes)
    g1[1] = (int)((tensor_w & 0xFFFFu) << 16);         // tensor_dim0[15:0]
    g1[2] = (int)((tensor_w >> 16) |                   // tensor_dim0[31:16]
                  ((tensor_h & 0xFFFFu) << 16));       // tensor_dim1[15:0]
    g1[3] = (int)(((tensor_h >> 16) & 0xFFFFu) |       // tensor_dim1[31:16]
                  (tile_w << 16));                     // tile_dim0
    g1[4] = (int)tile_h;                               // tile_dim1
    g1[5] = (int)stride;                               // tensor_dim0_stride lo
    i32x4 z4 = {};
#if defined(__clang_major__) && (__clang_major__ >= 23)
    i32x8 z8 = {};
    __builtin_amdgcn_tensor_load_to_lds(g0, g1, z4, z4, z8, 0);
#else
    __builtin_amdgcn_tensor_load_to_lds(g0, g1, z4, z4, 0);
#endif
}
#endif // USE_TDM

// ------------------------------------------------------------------- GEMM ---
// C[M,N] = A[M,K](bf16,row) * W[N,K](bf16,row)^T, fp32 out.
// Block: 256 thr (8 waves). Tile: BM=128 x BN=64, K-step 32.
// TDM path: wave 0 drives double-buffered LDS fills via tensor_load_to_lds,
// synchronized with s_wait_tensorcnt + workgroup barrier; all 8 waves spend
// their slots purely on ds_load_b128 + v_wmma.
#define BM 128
#define BN 64
#define BK 32

__global__ __launch_bounds__(256)
void gemm_bf16_wmma(const unsigned short* __restrict__ A,
                    const unsigned short* __restrict__ W,
                    float* __restrict__ C, int M, int Nn, int K) {
    __shared__ __attribute__((aligned(16))) unsigned short At[2][BM * BK];
    __shared__ __attribute__((aligned(16))) unsigned short Wt[2][BN * BK];

    const int tid  = threadIdx.x;
    const int lane = tid & 31, wave = tid >> 5;
    const int ll = lane & 15, lh = lane >> 4;
    const int wm = (wave & 3) * 32;      // wave row offset in tile
    const int wn = (wave >> 2) * 32;     // wave col offset in tile
    const int m0 = blockIdx.x * BM;
    const int n0 = blockIdx.y * BN;

    v8f acc[2][2] = {};
    int buf = 0;

#ifdef USE_TDM
    const unsigned short* Abase = A + (size_t)m0 * K;
    const unsigned short* Wbase = W + (size_t)n0 * K;
    unsigned int atOff[2], wtOff[2];
    atOff[0] = (unsigned int)(unsigned long long)(void*)&At[0][0];
    atOff[1] = (unsigned int)(unsigned long long)(void*)&At[1][0];
    wtOff[0] = (unsigned int)(unsigned long long)(void*)&Wt[0][0];
    wtOff[1] = (unsigned int)(unsigned long long)(void*)&Wt[1][0];
    if (wave == 0) {                     // prologue fill of buffer 0
        tdm_load_tile_bf16(atOff[0], Abase, BK, BM, (unsigned)K, (unsigned)M, (unsigned)K);
        tdm_load_tile_bf16(wtOff[0], Wbase, BK, BN, (unsigned)K, (unsigned)Nn, (unsigned)K);
        __builtin_amdgcn_s_wait_tensorcnt(0);
    }
    __syncthreads();
#else
    const int arow = tid >> 1, acb = (tid & 1) * 16;   // A coop-load coords
    const int wrow = tid >> 2, wcb = (tid & 3) * 8;    // W coop-load coords
#endif

    for (int k0 = 0; k0 < K; k0 += BK) {
#ifdef USE_TDM
        if (wave == 0 && k0 + BK < K) {  // async fill of the other buffer
            tdm_load_tile_bf16(atOff[buf ^ 1], Abase + k0 + BK,
                               BK, BM, (unsigned)K, (unsigned)M, (unsigned)K);
            tdm_load_tile_bf16(wtOff[buf ^ 1], Wbase + k0 + BK,
                               BK, BN, (unsigned)K, (unsigned)Nn, (unsigned)K);
        }
#else
        {   // cooperative global -> LDS staging (16B vector loads)
            const v4u* ga = (const v4u*)(A + (size_t)(m0 + arow) * K + k0 + acb);
            *(v4u*)(At[0] + arow * BK + acb)     = ga[0];
            *(v4u*)(At[0] + arow * BK + acb + 8) = ga[1];
            *(v4u*)(Wt[0] + wrow * BK + wcb) =
                *(const v4u*)(W + (size_t)(n0 + wrow) * K + k0 + wcb);
            if (k0 + BK < K) {   // emits global_prefetch_b8
                __builtin_prefetch(A + (size_t)(m0 + arow) * K + k0 + BK + acb, 0, 0);
                __builtin_prefetch(W + (size_t)(n0 + wrow) * K + k0 + BK + wcb, 0, 0);
            }
        }
        __syncthreads();
#endif
        const unsigned short* Ab = At[buf];
        const unsigned short* Wb = Wt[buf];

        FragBF a[2], b[2];
        #pragma unroll
        for (int mi = 0; mi < 2; ++mi) {        // A 16x32 frag (ISA layout)
            int row = wm + mi * 16 + ll;
            a[mi].u2[0] = *(const v4u*)(Ab + row * BK + lh * 8);
            a[mi].u2[1] = *(const v4u*)(Ab + row * BK + 16 + lh * 8);
        }
        #pragma unroll
        for (int ni = 0; ni < 2; ++ni) {        // B 32x16 frag: col=lane%16
            int nrow = wn + ni * 16 + ll;
            b[ni].u2[0] = *(const v4u*)(Wb + nrow * BK + lh * 8);
            b[ni].u2[1] = *(const v4u*)(Wb + nrow * BK + 16 + lh * 8);
        }
        #pragma unroll
        for (int mi = 0; mi < 2; ++mi)
            #pragma unroll
            for (int ni = 0; ni < 2; ++ni)
                acc[mi][ni] = __builtin_amdgcn_wmma_f32_16x16x32_bf16(
                    false, a[mi].bf, false, b[ni].bf,
                    (short)0, acc[mi][ni], false, false);

#ifdef USE_TDM
        if (wave == 0) __builtin_amdgcn_s_wait_tensorcnt(0);
        __syncthreads();                 // publish next buffer / release cur
        buf ^= 1;
#else
        __syncthreads();
#endif
    }

    #pragma unroll
    for (int mi = 0; mi < 2; ++mi)
        #pragma unroll
        for (int ni = 0; ni < 2; ++ni)
            #pragma unroll
            for (int r = 0; r < 8; ++r) {
                int row = m0 + wm + mi * 16 + r + lh * 8;
                int col = n0 + wn + ni * 16 + ll;
                C[(size_t)row * Nn + col] = acc[mi][ni][r];
            }
}

// ------------------------------------------------------------------- RoPE ---
// fused fp32 [NB, LQ, 73, 64] -> Q bf16 [NB, H, LQ, 64] (scaled by 1/8),
// K bf16 [NB, LQ, 64], V bf16 [NB, LQ, 64].
__global__ void rope_split(const float* __restrict__ fused,
                           unsigned short* __restrict__ Qb,
                           unsigned short* __restrict__ Kb,
                           unsigned short* __restrict__ Vb) {
    size_t idx = (size_t)blockIdx.x * blockDim.x + threadIdx.x;
    const size_t total = (size_t)NB * LQ * 73 * DKV;
    if (idx >= total) return;
    int d = idx & 63;
    size_t t1 = idx >> 6;
    int head = (int)(t1 % 73);
    size_t t2 = t1 / 73;
    int l  = (int)(t2 % LQ);
    int nb = (int)(t2 / LQ);
    size_t row = (size_t)nb * LQ + l;
    const float* frow = fused + row * NQKV + (size_t)head * DKV;
    float x = frow[d];
    if (head == 72) { Vb[row * DKV + d] = f2bf(x); return; }
    int fi = d & 31;
    float inv = __powf(10000.0f, -(float)(2 * fi) * (1.0f / 64.0f));
    float ang = (float)l * inv;
    float rot = (d < 32) ? -frow[d + 32] : frow[d - 32];
    float out = x * __cosf(ang) + rot * __sinf(ang);
    if (head == 71)
        Kb[row * DKV + d] = f2bf(out);
    else
        Qb[((size_t)(nb * NHEAD + head) * LQ + l) * DKV + d] = f2bf(out * 0.125f);
}

// -------------------------------------------------------------- attention ---
// One wave per (query-tile of 16, head, batch). Flash-style online softmax.
__global__ __launch_bounds__(32)
void attn_fwd(const unsigned short* __restrict__ Qb,
              const unsigned short* __restrict__ Kb,
              const unsigned short* __restrict__ Vb,
              unsigned short* __restrict__ Ao) {
    __shared__ __attribute__((aligned(16))) unsigned short Plds[16 * 32];
    __shared__ __attribute__((aligned(16))) unsigned short Vlds[32 * DKV];

    const int lane = threadIdx.x, ll = lane & 15, lh = lane >> 4;
    const int qt = blockIdx.x, h = blockIdx.y, nb = blockIdx.z;
    const int qBase = qt * 16;

    const unsigned short* Qp = Qb + (size_t)(nb * NHEAD + h) * LQ * DKV;
    const unsigned short* Kp = Kb + (size_t)nb * LQ * DKV;
    const unsigned short* Vp = Vb + (size_t)nb * LQ * DKV;

    FragBF qf[2];
    {
        const unsigned short* qr = Qp + (size_t)(qBase + ll) * DKV;
        qf[0].u2[0] = *(const v4u*)(qr + lh * 8);
        qf[0].u2[1] = *(const v4u*)(qr + 16 + lh * 8);
        qf[1].u2[0] = *(const v4u*)(qr + 32 + lh * 8);
        qf[1].u2[1] = *(const v4u*)(qr + 48 + lh * 8);
    }

    float mrun[8], lsum[8];
    v8f Oacc[4] = {};
    #pragma unroll
    for (int r = 0; r < 8; ++r) { mrun[r] = -3.0e38f; lsum[r] = 0.0f; }

    const int nkb = (qBase + 16 + 31) >> 5;     // causal: key blocks of 32
    for (int jb = 0; jb < nkb; ++jb) {
        const int kb = jb * 32;
        // stage 32 V rows into LDS (128B contiguous per lane)
        {
            const v4u* vg = (const v4u*)(Vp + (size_t)(kb + lane) * DKV);
            #pragma unroll
            for (int i = 0; i < 8; ++i)
                *(v4u*)(Vlds + lane * DKV + i * 8) = vg[i];
        }
        // S(16x32) = Q(16x64) * K^T  -> 4 WMMA
        v8f s[2] = {};
        #pragma unroll
        for (int nt = 0; nt < 2; ++nt) {
            const unsigned short* kr = Kp + (size_t)(kb + nt * 16 + ll) * DKV;
            #pragma unroll
            for (int dh = 0; dh < 2; ++dh) {
                FragBF bk;
                bk.u2[0] = *(const v4u*)(kr + dh * 32 + lh * 8);
                bk.u2[1] = *(const v4u*)(kr + dh * 32 + 16 + lh * 8);
                s[nt] = __builtin_amdgcn_wmma_f32_16x16x32_bf16(
                    false, qf[dh].bf, false, bk.bf, (short)0, s[nt], false, false);
            }
        }
        // branch-free causal mask + online softmax (rows = (r, lane-half))
        float scale[8];
        #pragma unroll
        for (int r = 0; r < 8; ++r) {
            int qRow = qBase + r + lh * 8;
            if (kb + ll > qRow)      s[0][r] = -3.0e38f;
            if (kb + 16 + ll > qRow) s[1][r] = -3.0e38f;
            float c = fmaxf(s[0][r], s[1][r]);
            #pragma unroll
            for (int m = 1; m < 16; m <<= 1) c = fmaxf(c, __shfl_xor(c, m, 32));
            float mnew = fmaxf(mrun[r], c);
            scale[r] = __expf(mrun[r] - mnew);
            float p0 = __expf(s[0][r] - mnew);
            float p1 = __expf(s[1][r] - mnew);
            float rs = p0 + p1;
            #pragma unroll
            for (int m = 1; m < 16; m <<= 1) rs += __shfl_xor(rs, m, 32);
            lsum[r] = lsum[r] * scale[r] + rs;
            mrun[r] = mnew;
            Plds[(r + lh * 8) * 32 + ll]      = f2bf(p0);
            Plds[(r + lh * 8) * 32 + 16 + ll] = f2bf(p1);
        }
        #pragma unroll
        for (int t = 0; t < 4; ++t)
            #pragma unroll
            for (int r = 0; r < 8; ++r) Oacc[t][r] *= scale[r];

        // P (C-layout -> A-fragment) via LDS round trip
        FragBF pf;
        pf.u2[0] = *(const v4u*)(Plds + ll * 32 + lh * 8);
        pf.u2[1] = *(const v4u*)(Plds + ll * 32 + 16 + lh * 8);
        // O(16x64) += P(16x32) * V(32x64)  -> 4 WMMA
        #pragma unroll
        for (int t = 0; t < 4; ++t) {
            FragBF vf;
            #pragma unroll
            for (int kk = 0; kk < 16; ++kk)
                vf.s[kk] = Vlds[((kk >> 3) * 16 + lh * 8 + (kk & 7)) * DKV + t * 16 + ll];
            Oacc[t] = __builtin_amdgcn_wmma_f32_16x16x32_bf16(
                false, pf.bf, false, vf.bf, (short)0, Oacc[t], false, false);
        }
    }

    // normalize and write bf16 [NB*LQ, H*64]
    #pragma unroll
    for (int r = 0; r < 8; ++r) {
        int qRow = qBase + r + lh * 8;
        float inv = 1.0f / lsum[r];
        size_t base = ((size_t)nb * LQ + qRow) * (size_t)(NHEAD * DKV) + (size_t)h * DKV;
        #pragma unroll
        for (int t = 0; t < 4; ++t)
            Ao[base + t * 16 + ll] = f2bf(Oacc[t][r] * inv);
    }
}

// ------------------------------------------------------------------- host ---
extern "C" void kernel_launch(void* const* d_in, const int* in_sizes, int n_in,
                              void* d_out, int out_size, void* d_ws, size_t ws_size,
                              hipStream_t stream) {
    const float* hidden  = (const float*)d_in[0];   // [NB, LQ, D]
    const float* w_qkv   = (const float*)d_in[1];   // [NQKV, D]
    const float* w_dense = (const float*)d_in[2];   // [D, D]
    float* out = (float*)d_out;                     // [NB, LQ, D]

    // workspace carve-out (~179 MB total)
    char* w = (char*)d_ws;
    auto carve = [&](size_t bytes) {
        void* p = (void*)w;
        w += (bytes + 255) & ~(size_t)255;
        return p;
    };
    const size_t nHid = (size_t)MROWS * DMODEL;
    const size_t nQkvW = (size_t)NQKV * DMODEL;
    const size_t nDenW = (size_t)DMODEL * DMODEL;
    unsigned short* hb  = (unsigned short*)carve(nHid * 2);
    unsigned short* wq  = (unsigned short*)carve(nQkvW * 2);
    unsigned short* wd  = (unsigned short*)carve(nDenW * 2);
    float* fused        = (float*)carve((size_t)MROWS * NQKV * 4);
    unsigned short* Qb  = (unsigned short*)carve((size_t)NB * NHEAD * LQ * DKV * 2);
    unsigned short* Kb  = (unsigned short*)carve((size_t)NB * LQ * DKV * 2);
    unsigned short* Vb  = (unsigned short*)carve((size_t)NB * LQ * DKV * 2);
    unsigned short* Ao  = (unsigned short*)carve(nHid * 2);

    // 1) fp32 -> bf16 conversions
    cvt_f32_bf16<<<(int)((nHid  + 255) / 256), 256, 0, stream>>>(hidden,  hb, nHid);
    cvt_f32_bf16<<<(int)((nQkvW + 255) / 256), 256, 0, stream>>>(w_qkv,   wq, nQkvW);
    cvt_f32_bf16<<<(int)((nDenW + 255) / 256), 256, 0, stream>>>(w_dense, wd, nDenW);

    // 2) fused QKV GEMM: [2048,4544] x [4672,4544]^T -> fp32 [2048,4672]
    gemm_bf16_wmma<<<dim3(MROWS / BM, NQKV / BN), 256, 0, stream>>>(
        hb, wq, fused, MROWS, NQKV, DMODEL);

    // 3) RoPE + split into Q/K/V (scale 1/sqrt(64) folded into Q)
    {
        size_t total = (size_t)NB * LQ * 73 * DKV;
        rope_split<<<(int)((total + 255) / 256), 256, 0, stream>>>(fused, Qb, Kb, Vb);
    }

    // 4) causal flash attention: one wave per (16 queries, head, batch)
    attn_fwd<<<dim3(LQ / 16, NHEAD, NB), 32, 0, stream>>>(Qb, Kb, Vb, Ao);

    // 5) dense projection: [2048,4544] x [4544,4544]^T -> fp32 out
    gemm_bf16_wmma<<<dim3(MROWS / BM, DMODEL / BN), 256, 0, stream>>>(
        Ao, wd, out, MROWS, DMODEL, DMODEL);
}
